// MultiHeadAttentionLayer_82497731822091
// MI455X (gfx1250) — matrix operand order repaired and checked by reading, not compiled, not run on hardware
//
#include <hip/hip_runtime.h>
#include <hip/hip_bf16.h>
#include <math.h>

#define N_NODES 50000
#define N_EDGES 800000
#define IN_DIM  128
#define N_HEADS 8
#define D_HEAD  16
#define OUT_DIM 128   /* N_HEADS * D_HEAD */

#define MTILES_PER_BLOCK 8        /* 8 x 16 = 128 rows of x per block */
#define N_MTILES (N_NODES / 16)   /* 3125, exact */

typedef __attribute__((ext_vector_type(16))) _Float16 v16h;
typedef __attribute__((ext_vector_type(8)))  float    v8f;

// ---------------------------------------------------------------------------
// Kernel 1: fused QKV projection GEMM via V_WMMA_F32_16X16X32_F16.
// grid = (ceil(3125/8), 3), block = 256 (8 waves). Wave w owns output columns
// [w*16, +16) of matrix blockIdx.y (0=Q,1=K,2=V). It preloads the four
// 32x16 B fragments (full K=128) ONCE into registers, then sweeps 8 M-tiles
// reusing them -> strided W loads amortized 8x, inner loop is A-loads + wmma.
// Fragment layouts used here (wave32; A and B share contiguous k order):
//   A 16x32 f16 : lane L holds row (L&15); lanes<16 pack K = k0+{0..15},
//                 lanes>=16 pack K = k0+{16..31}.
//   B 32x16 f16 : lane L holds col (L&15); VGPR j holds K = k0+(L<16?0:16)+j.
//   C/D 16x16 f32: lane L col (L&15); acc[r] is row r + (L<16?0:8).
// ---------------------------------------------------------------------------
__global__ __launch_bounds__(256)
void qkv_gemm_wmma(const float* __restrict__ x,
                   const float* __restrict__ Wq, const float* __restrict__ bq,
                   const float* __restrict__ Wk, const float* __restrict__ bk,
                   const float* __restrict__ Wv, const float* __restrict__ bv,
                   float* __restrict__ Q, float* __restrict__ K,
                   float* __restrict__ V)
{
    const int mat = blockIdx.y;
    const float* W;
    const float* b;
    float* out;
    if (mat == 0)      { W = Wq; b = bq; out = Q; }
    else if (mat == 1) { W = Wk; b = bk; out = K; }
    else               { W = Wv; b = bv; out = V; }

    const int wave = threadIdx.x >> 5;    // 0..7 -> N tile
    const int lane = threadIdx.x & 31;
    const int hi   = lane >> 4;           // 0: lanes 0-15, 1: lanes 16-31
    const int col  = wave * 16 + (lane & 15);

    // ---- preload all four B fragments (K = 0..127), reused for 8 M-tiles ----
    v16h bf[4];
    #pragma unroll
    for (int kk = 0; kk < 4; ++kk) {
        const float* bp = W + (size_t)(kk * 32 + (hi ? 16 : 0)) * OUT_DIM + col;
        #pragma unroll
        for (int j = 0; j < 16; ++j) bf[kk][j] = (_Float16)bp[j * OUT_DIM];
    }
    const float bias = b[col];

    const int tile0 = blockIdx.x * MTILES_PER_BLOCK;
    #pragma unroll 1
    for (int t = 0; t < MTILES_PER_BLOCK; ++t) {
        const int mt = tile0 + t;
        if (mt >= N_MTILES) break;            // block-uniform: EXEC stays full
        const int m0   = mt * 16;
        const int arow = m0 + (lane & 15);

        v8f acc = {};
        #pragma unroll
        for (int kk = 0; kk < 4; ++kk) {
            v16h a;
            const float* ap = x + (size_t)arow * IN_DIM + kk * 32 + (hi ? 16 : 0);
            #pragma unroll
            for (int j = 0; j < 8; ++j) {
                a[j]     = (_Float16)ap[j];
                a[8 + j] = (_Float16)ap[8 + j];
            }
            acc = __builtin_amdgcn_wmma_f32_16x16x32_f16(
                      /*neg_a=*/false, a, /*neg_b=*/false, bf[kk],
                      /*c_mod=*/(short)0, acc,
                      /*reuse_a=*/false, /*reuse_b=*/false);
        }

        #pragma unroll
        for (int r = 0; r < 8; ++r) {
            const int m = m0 + r + hi * 8;
            out[(size_t)m * OUT_DIM + col] = acc[r] + bias;
        }
    }
}

// ---------------------------------------------------------------------------
// Kernel 0: init segment buffers (seg_max = -inf bits, seg_sum = 0).
// ---------------------------------------------------------------------------
__global__ void seg_init_kernel(unsigned int* __restrict__ seg_max_bits,
                                float* __restrict__ seg_sum, int n)
{
    const int t = blockIdx.x * blockDim.x + threadIdx.x;
    if (t < n) {
        seg_max_bits[t] = 0xFF800000u;   // -inf
        seg_sum[t] = 0.0f;
    }
}

// ---------------------------------------------------------------------------
// Kernel 2: per-(edge,head) attention score + segment max.
// score = clip(dot(Q[dst,h,:], K[src,h,:]) / 4, -5, 5)
// Float atomic-max via sign-aware int/uint punning (init = -inf works).
// ---------------------------------------------------------------------------
__global__ void edge_score_kernel(const float* __restrict__ Q,
                                  const float* __restrict__ Kp,
                                  const int* __restrict__ src,
                                  const int* __restrict__ dst,
                                  float* __restrict__ score,
                                  unsigned int* __restrict__ seg_max_bits)
{
    const int t = blockIdx.x * blockDim.x + threadIdx.x;
    if (t >= N_EDGES * N_HEADS) return;
    const int e = t >> 3;
    const int h = t & 7;
    const int s = src[e];
    const int d = dst[e];

    const float4* q = (const float4*)(Q  + (size_t)d * OUT_DIM + h * D_HEAD);
    const float4* k = (const float4*)(Kp + (size_t)s * OUT_DIM + h * D_HEAD);
    float acc = 0.0f;
    #pragma unroll
    for (int i = 0; i < 4; ++i) {
        const float4 qa = q[i];
        const float4 kb = k[i];
        acc += qa.x * kb.x + qa.y * kb.y + qa.z * kb.z + qa.w * kb.w;
    }
    acc *= 0.25f;                                   // 1/sqrt(D_HEAD)
    acc = fminf(5.0f, fmaxf(-5.0f, acc));
    score[t] = acc;

    unsigned int* addr = seg_max_bits + (size_t)d * N_HEADS + h;
    const unsigned int ub = __float_as_uint(acc);
    if ((int)ub >= 0) atomicMax((int*)addr, (int)ub);   // value >= 0
    else              atomicMin(addr, ub);              // value <  0
}

// ---------------------------------------------------------------------------
// Kernel 3: e = exp(score - seg_max[dst]); seg_sum[dst] += e (in place).
// ---------------------------------------------------------------------------
__global__ void edge_exp_kernel(const int* __restrict__ dst,
                                const float* __restrict__ seg_max,
                                float* __restrict__ score,
                                float* __restrict__ seg_sum)
{
    const int t = blockIdx.x * blockDim.x + threadIdx.x;
    if (t >= N_EDGES * N_HEADS) return;
    const int e = t >> 3;
    const int h = t & 7;
    const int d = dst[e];
    const float m  = seg_max[(size_t)d * N_HEADS + h];
    const float ev = __expf(score[t] - m);
    score[t] = ev;                                   // reuse buffer for e
    atomicAdd(seg_sum + (size_t)d * N_HEADS + h, ev);
}

// ---------------------------------------------------------------------------
// Kernel 4: out[dst] += (e/seg_sum[dst]) * V[src].  One wave per edge,
// lane l handles floats [4l, 4l+4) of the 128-wide row (head = l>>2).
// ---------------------------------------------------------------------------
__global__ __launch_bounds__(256)
void edge_scatter_kernel(const float* __restrict__ V,
                         const int* __restrict__ src,
                         const int* __restrict__ dst,
                         const float* __restrict__ e_val,
                         const float* __restrict__ seg_sum,
                         float* __restrict__ out)
{
    const int lane = threadIdx.x & 31;
    const int eidx = blockIdx.x * (blockDim.x >> 5) + (threadIdx.x >> 5);
    if (eidx >= N_EDGES) return;
    const int s = src[eidx];
    const int d = dst[eidx];
    const int h = lane >> 2;

    const float alpha = e_val[(size_t)eidx * N_HEADS + h] /
                        seg_sum[(size_t)d * N_HEADS + h];
    const float4 v = ((const float4*)(V + (size_t)s * OUT_DIM))[lane];
    float* op = out + (size_t)d * OUT_DIM + lane * 4;
    atomicAdd(op + 0, alpha * v.x);
    atomicAdd(op + 1, alpha * v.y);
    atomicAdd(op + 2, alpha * v.z);
    atomicAdd(op + 3, alpha * v.w);
}

// ---------------------------------------------------------------------------
extern "C" void kernel_launch(void* const* d_in, const int* in_sizes, int n_in,
                              void* d_out, int out_size, void* d_ws, size_t ws_size,
                              hipStream_t stream)
{
    (void)in_sizes; (void)n_in; (void)ws_size;

    const float* x  = (const float*)d_in[0];
    const int*   ei = (const int*)  d_in[1];     // [2, E] flat
    const float* Wq = (const float*)d_in[2];
    const float* bq = (const float*)d_in[3];
    const float* Wk = (const float*)d_in[4];
    const float* bk = (const float*)d_in[5];
    const float* Wv = (const float*)d_in[6];
    const float* bv = (const float*)d_in[7];
    float* out = (float*)d_out;

    const int* src = ei;
    const int* dst = ei + N_EDGES;

    // workspace carve-up (~106 MB)
    float* Q       = (float*)d_ws;
    float* K       = Q       + (size_t)N_NODES * OUT_DIM;
    float* V       = K       + (size_t)N_NODES * OUT_DIM;
    float* score   = V       + (size_t)N_NODES * OUT_DIM;   // E*H, reused as e
    float* seg_max = score   + (size_t)N_EDGES * N_HEADS;
    float* seg_sum = seg_max + (size_t)N_NODES * N_HEADS;

    // zero output accumulator every call (graph-capture safe)
    hipMemsetAsync(d_out, 0, (size_t)out_size * sizeof(float), stream);

    const int nseg = N_NODES * N_HEADS;
    seg_init_kernel<<<(nseg + 255) / 256, 256, 0, stream>>>(
        (unsigned int*)seg_max, seg_sum, nseg);

    dim3 gemm_grid((N_MTILES + MTILES_PER_BLOCK - 1) / MTILES_PER_BLOCK, 3);
    qkv_gemm_wmma<<<gemm_grid, 256, 0, stream>>>(
        x, Wq, bq, Wk, bk, Wv, bv, Q, K, V);

    const int eh = N_EDGES * N_HEADS;
    edge_score_kernel<<<(eh + 255) / 256, 256, 0, stream>>>(
        Q, K, src, dst, score, (unsigned int*)seg_max);

    edge_exp_kernel<<<(eh + 255) / 256, 256, 0, stream>>>(
        dst, seg_max, score, seg_sum);

    edge_scatter_kernel<<<(N_EDGES + 7) / 8, 256, 0, stream>>>(
        V, src, dst, score, seg_sum, out);
}
